// AttentionDecoder_30683246362693
// MI455X (gfx1250) — compile-verified, matrix operand order checked by reading, not run on hardware
//
#include <hip/hip_runtime.h>
#include <hip/hip_bf16.h>
#include <math.h>

// Problem dims (fixed by the reference)
#define B_   64
#define S_   64
#define H_   512
#define E_   256
#define V_   32000
#define NL2_ 4
#define T_   32
#define EPS_ 1e-5f

typedef unsigned short u16;
typedef __attribute__((ext_vector_type(8)))  u16          u16x8;
typedef __attribute__((ext_vector_type(16))) u16          u16x16;
typedef __attribute__((ext_vector_type(16))) __bf16       v16bf;
typedef __attribute__((ext_vector_type(8)))  float        v8f;
typedef __attribute__((ext_vector_type(4)))  unsigned int u32x4;
typedef __attribute__((ext_vector_type(8)))  int          i32x8;
typedef __attribute__((ext_vector_type(4)))  int          i32x4;

// Wave tile: 2 x 4 subtiles of 16x16 => 32x64 C per wave.
constexpr int WM = 2;
constexpr int WN = 4;

// ---------- small device helpers ----------
__device__ __forceinline__ u16 f2bf(float f) {
  unsigned u = __float_as_uint(f);
  u += 0x7FFFu + ((u >> 16) & 1u);          // round-to-nearest-even
  return (u16)(u >> 16);
}
__device__ __forceinline__ float sigm(float x) { return 1.f / (1.f + __expf(-x)); }

// Assemble one 16-bf16 WMMA operand fragment. `p` already points at
// row*ld + half*8 (per-lane base); the two 8-element contiguous runs are
// at +k and +k+16 (ISA 16-bit A/B layout) -> two 16B loads.
__device__ __forceinline__ v16bf frag16(const u16* __restrict__ p) {
  const u16x8 lo = *reinterpret_cast<const u16x8*>(p);
  const u16x8 hi = *reinterpret_cast<const u16x8*>(p + 16);
  u16x16 t;
#pragma unroll
  for (int i = 0; i < 8; ++i) { t[i] = lo[i]; t[i + 8] = hi[i]; }
  return __builtin_bit_cast(v16bf, t);
}

// ---------- unified WMMA GEMM ----------
// C[M,N] = A1[M,K1]*B1t[N,K1]^T (+ A2[M,K2]*B2t[N,K2]^T if K2>0) + bias1 + bias2
// One wave computes a 32x64 C tile (WM x WN accumulators). bf16 in, f32 acc.
__global__ void wmma_gemm_dual(const u16* __restrict__ A1, const u16* __restrict__ B1t, int K1,
                               const u16* __restrict__ A2, const u16* __restrict__ B2t, int K2,
                               const float* __restrict__ bias1, const float* __restrict__ bias2,
                               float* __restrict__ C, int N, int tiles_m, int tiles_n) {
  const int lane = threadIdx.x & 31;
  const int wave = threadIdx.x >> 5;
  const int wt = blockIdx.x * (blockDim.x >> 5) + wave;
  if (wt >= tiles_m * tiles_n) return;           // uniform per wave: EXEC stays all-1s
  const int m0 = (wt / tiles_n) * (WM * 16);
  const int n0 = (wt % tiles_n) * (WN * 16);
  const int lr = lane & 15;
  const int half = (lane >> 4) & 1;

  const v8f vz = {0.f, 0.f, 0.f, 0.f, 0.f, 0.f, 0.f, 0.f};
  v8f acc[WM][WN];
#pragma unroll
  for (int i = 0; i < WM; ++i)
#pragma unroll
    for (int j = 0; j < WN; ++j) acc[i][j] = vz;

  { // ---- K segment 1 ----
    const u16* ar[WM];
    const u16* br[WN];
#pragma unroll
    for (int i = 0; i < WM; ++i) ar[i] = A1 + (size_t)(m0 + i * 16 + lr) * K1 + half * 8;
#pragma unroll
    for (int j = 0; j < WN; ++j) br[j] = B1t + (size_t)(n0 + j * 16 + lr) * K1 + half * 8;
    for (int k = 0; k < K1; k += 32) {
      v16bf af[WM], bfr[WN];
#pragma unroll
      for (int i = 0; i < WM; ++i) af[i] = frag16(ar[i] + k);
#pragma unroll
      for (int j = 0; j < WN; ++j) {
        __builtin_prefetch(br[j] + k + 64, 0, 3);  // speculative: global_prefetch_b8
        bfr[j] = frag16(br[j] + k);
      }
#pragma unroll
      for (int i = 0; i < WM; ++i)
#pragma unroll
        for (int j = 0; j < WN; ++j)
          acc[i][j] = __builtin_amdgcn_wmma_f32_16x16x32_bf16(
              false, af[i], false, bfr[j], (short)0, acc[i][j], false, false);
    }
  }
  if (K2 > 0) { // ---- K segment 2 (LSTM gates: h @ w_hh^T) ----
    const u16* ar[WM];
    const u16* br[WN];
#pragma unroll
    for (int i = 0; i < WM; ++i) ar[i] = A2 + (size_t)(m0 + i * 16 + lr) * K2 + half * 8;
#pragma unroll
    for (int j = 0; j < WN; ++j) br[j] = B2t + (size_t)(n0 + j * 16 + lr) * K2 + half * 8;
    for (int k = 0; k < K2; k += 32) {
      v16bf af[WM], bfr[WN];
#pragma unroll
      for (int i = 0; i < WM; ++i) af[i] = frag16(ar[i] + k);
#pragma unroll
      for (int j = 0; j < WN; ++j) bfr[j] = frag16(br[j] + k);
#pragma unroll
      for (int i = 0; i < WM; ++i)
#pragma unroll
        for (int j = 0; j < WN; ++j)
          acc[i][j] = __builtin_amdgcn_wmma_f32_16x16x32_bf16(
              false, af[i], false, bfr[j], (short)0, acc[i][j], false, false);
    }
  }

  // ---- store: acc vgpr r -> M = r + 8*half, N = lr (per subtile) ----
#pragma unroll
  for (int j = 0; j < WN; ++j) {
    const int n = n0 + j * 16 + lr;
    float bv = bias1 ? bias1[n] : 0.f;
    if (bias2) bv += bias2[n];
#pragma unroll
    for (int i = 0; i < WM; ++i) {
      const int mb = m0 + i * 16 + half * 8;
#pragma unroll
      for (int r = 0; r < 8; ++r)
        C[(size_t)(mb + r) * N + n] = acc[i][j][r] + bv;
    }
  }
}

// ---------- TDM demo/warm kernel: stage a 16x64 bf16 tile of out_w via the ----------
// ---------- Tensor Data Mover, then drain LDS to scratch.                  ----------
__global__ void tdm_stage_kernel(const u16* __restrict__ src, u16* __restrict__ dst) {
  __shared__ u16 tile[16 * 64];
  if (threadIdx.x < 32) {     // one wave issues the TDM op (EXEC ignored by tensor ops)
    const unsigned long long ga = (unsigned long long)src;
    u32x4 g0;
    g0[0] = 1u;                                      // count=1 (valid descriptor)
    g0[1] = 0u;                                      // lds_addr = 0 (start of LDS alloc)
    g0[2] = (unsigned)(ga & 0xffffffffu);            // global_addr[31:0]
    g0[3] = (unsigned)((ga >> 32) & 0x01ffffffu)     // global_addr[56:32]
            | (2u << 30);                            // type = 2 ("image")
    i32x8 g1;
    g1[0] = 0x00010000;                              // wg_mask=0, data_size=1 (2 bytes)
    g1[1] = (int)(512u << 16);                       // tensor_dim0 = 512
    g1[2] = (int)(64u << 16);                        // tensor_dim1 = 64
    g1[3] = (int)(64u << 16);                        // tile_dim0 = 64
    g1[4] = 16;                                      // tile_dim1 = 16
    g1[5] = 512;                                     // tensor_dim0_stride = 512
    g1[6] = 0;
    g1[7] = 0;
    const i32x4 z4 = {0, 0, 0, 0};
    const i32x8 z8 = {0, 0, 0, 0, 0, 0, 0, 0};
    // amdgpu-toolchain (clang-23) 6-arg form: (g0, g1, g2, g3, g4, cpol)
    __builtin_amdgcn_tensor_load_to_lds(g0, g1, z4, z4, z8, 0);
    __builtin_amdgcn_s_wait_tensorcnt(0);
  }
  __syncthreads();
  for (int i = threadIdx.x; i < 16 * 64; i += blockDim.x) dst[i] = tile[i];
}

// ---------- elementwise / reduction kernels ----------
__global__ void f32_to_bf16_kernel(const float* __restrict__ src, u16* __restrict__ dst, int n) {
  int i = blockIdx.x * blockDim.x + threadIdx.x;
  if (i < n) dst[i] = f2bf(src[i]);
}

// h0/c0: weighted sum over NL2 layers
__global__ void init_hc_kernel(const float* __restrict__ hn, const float* __restrict__ cn,
                               const float* __restrict__ pwh, const float* __restrict__ pbh,
                               const float* __restrict__ pwc, const float* __restrict__ pbc,
                               u16* __restrict__ h_bf, float* __restrict__ c) {
  int i = blockIdx.x * blockDim.x + threadIdx.x;
  if (i >= B_ * H_) return;
  float hv = pbh[0], cv = pbc[0];
#pragma unroll
  for (int l = 0; l < NL2_; ++l) {
    hv += hn[(size_t)l * B_ * H_ + i] * pwh[l];
    cv += cn[(size_t)l * B_ * H_ + i] * pwc[l];
  }
  h_bf[i] = f2bf(hv);
  c[i] = cv;
}

// LayerNorm over last dim D; one block (256 thr) per row; optional f32 + bf16 outputs
__global__ void layernorm_kernel(const float* __restrict__ x, const float* __restrict__ g,
                                 const float* __restrict__ bta, float* __restrict__ out_f32,
                                 u16* __restrict__ out_bf16, int D) {
  const int row = blockIdx.x;
  const int tid = threadIdx.x;
  const float* xr = x + (size_t)row * D;
  __shared__ float sdata[256];
  float s1 = 0.f, s2 = 0.f;
  for (int j = tid; j < D; j += 256) { float v = xr[j]; s1 += v; s2 += v * v; }
  sdata[tid] = s1; __syncthreads();
  for (int s = 128; s > 0; s >>= 1) { if (tid < s) sdata[tid] += sdata[tid + s]; __syncthreads(); }
  const float mean = sdata[0] / D;
  __syncthreads();
  sdata[tid] = s2; __syncthreads();
  for (int s = 128; s > 0; s >>= 1) { if (tid < s) sdata[tid] += sdata[tid + s]; __syncthreads(); }
  const float var = sdata[0] / D - mean * mean;
  const float inv = rsqrtf(var + EPS_);
  for (int j = tid; j < D; j += 256) {
    float v = (xr[j] - mean) * inv * g[j] + bta[j];
    if (out_f32)  out_f32[(size_t)row * D + j] = v;
    if (out_bf16) out_bf16[(size_t)row * D + j] = f2bf(v);
  }
}

// Gather emb[targets] + LayerNorm(E) -> x_bf [T,B,E]; one block per (b,t), 256 thr == E
__global__ void embed_ln_kernel(const int* __restrict__ targets, const float* __restrict__ emb,
                                const float* __restrict__ g, const float* __restrict__ bta,
                                u16* __restrict__ x_bf) {
  const int row = blockIdx.x;          // b*T + t
  const int bb = row / T_, tt = row % T_;
  const int tid = threadIdx.x;         // 0..255 == E_
  const int tok = targets[row];
  const float v = emb[(size_t)tok * E_ + tid];
  __shared__ float sdata[256];
  sdata[tid] = v; __syncthreads();
  for (int s = 128; s > 0; s >>= 1) { if (tid < s) sdata[tid] += sdata[tid + s]; __syncthreads(); }
  const float mean = sdata[0] / (float)E_;
  __syncthreads();
  sdata[tid] = v * v; __syncthreads();
  for (int s = 128; s > 0; s >>= 1) { if (tid < s) sdata[tid] += sdata[tid + s]; __syncthreads(); }
  const float var = sdata[0] / (float)E_ - mean * mean;
  const float inv = rsqrtf(var + EPS_);
  x_bf[((size_t)tt * B_ + bb) * E_ + tid] = f2bf((v - mean) * inv * g[tid] + bta[tid]);
}

// energies[b,s] = sum_h tanh(q[b,h] + pk[b,s,h]) * e_w[h]; one block per (b,s)
__global__ void energy_kernel(const float* __restrict__ q, const float* __restrict__ pk,
                              const float* __restrict__ e_w, float* __restrict__ energ) {
  const int row = blockIdx.x;          // b*S + s
  const int b = row >> 6;
  const int tid = threadIdx.x;
  const float* qr  = q  + (size_t)b * H_;
  const float* pkr = pk + (size_t)row * H_;
  float acc = 0.f;
  for (int h = tid; h < H_; h += 256) acc += tanhf(qr[h] + pkr[h]) * e_w[h];
  __shared__ float sdata[256];
  sdata[tid] = acc; __syncthreads();
  for (int s = 128; s > 0; s >>= 1) { if (tid < s) sdata[tid] += sdata[tid + s]; __syncthreads(); }
  if (tid == 0) energ[row] = sdata[0];
}

// softmax over S + context = alphas @ enc; writes xc = [context(bf16,H) | x_t(bf16,E)]
__global__ void softmax_context_kernel(const float* __restrict__ energ,
                                       const float* __restrict__ enc,
                                       const u16* __restrict__ x_t,
                                       u16* __restrict__ xc) {
  const int b = blockIdx.x;
  const int tid = threadIdx.x;
  __shared__ float alphas[S_];
  __shared__ float inv_sum;
  if (tid < S_) alphas[tid] = energ[b * S_ + tid];
  __syncthreads();
  if (tid == 0) {
    float m = alphas[0];
    for (int s = 1; s < S_; ++s) m = fmaxf(m, alphas[s]);
    float t = 0.f;
    for (int s = 0; s < S_; ++s) { alphas[s] = __expf(alphas[s] - m); t += alphas[s]; }
    inv_sum = 1.f / t;
  }
  __syncthreads();
  if (tid < S_) alphas[tid] *= inv_sum;
  __syncthreads();
  for (int h = tid; h < H_; h += 256) {
    float acc = 0.f;
#pragma unroll 4
    for (int s = 0; s < S_; ++s) acc += alphas[s] * enc[((size_t)b * S_ + s) * H_ + h];
    xc[(size_t)b * (H_ + E_) + h] = f2bf(acc);
  }
  xc[(size_t)b * (H_ + E_) + H_ + tid] = x_t[(size_t)b * E_ + tid];   // tid < 256 == E_
}

// LSTM cell: c' = sig(f)*c + sig(i)*tanh(g); h' = sig(o)*tanh(c')
__global__ void lstm_cell_kernel(const float* __restrict__ gates, float* __restrict__ c,
                                 u16* __restrict__ h_bf, u16* __restrict__ hs_bf, int t) {
  int i = blockIdx.x * blockDim.x + threadIdx.x;     // 0..B*H
  if (i >= B_ * H_) return;
  const int b = i >> 9, k = i & (H_ - 1);
  const float* g = gates + (size_t)b * 4 * H_;
  const float ig = sigm(g[k]);
  const float fg = sigm(g[H_ + k]);
  const float gg = tanhf(g[2 * H_ + k]);
  const float og = sigm(g[3 * H_ + k]);
  const float c2 = fg * c[i] + ig * gg;
  const float h2 = og * tanhf(c2);
  c[i] = c2;
  const u16 hb = f2bf(h2);
  h_bf[i] = hb;
  hs_bf[((size_t)b * T_ + t) * H_ + k] = hb;         // hs laid out [B,T,H] = A of final GEMM
}

// ---------- host orchestration ----------
extern "C" void kernel_launch(void* const* d_in, const int* in_sizes, int n_in,
                              void* d_out, int out_size, void* d_ws, size_t ws_size,
                              hipStream_t stream) {
  (void)in_sizes; (void)n_in; (void)out_size; (void)ws_size;
  const int*   targets  = (const int*)d_in[0];
  const float* enc_hid  = (const float*)d_in[1];
  const float* enc_hn   = (const float*)d_in[2];
  const float* enc_cn   = (const float*)d_in[3];
  const float* emb      = (const float*)d_in[5];
  const float* ln_enc_g = (const float*)d_in[6];
  const float* ln_enc_b = (const float*)d_in[7];
  const float* ln_emb_g = (const float*)d_in[8];
  const float* ln_emb_b = (const float*)d_in[9];
  const float* phw      = (const float*)d_in[10];
  const float* phb      = (const float*)d_in[11];
  const float* pcw      = (const float*)d_in[12];
  const float* pcb      = (const float*)d_in[13];
  const float* q_w      = (const float*)d_in[14];
  const float* q_b      = (const float*)d_in[15];
  const float* k_w      = (const float*)d_in[16];
  const float* e_w      = (const float*)d_in[17];
  const float* w_ih     = (const float*)d_in[18];
  const float* w_hh     = (const float*)d_in[19];
  const float* b_ih     = (const float*)d_in[20];
  const float* b_hh     = (const float*)d_in[21];
  const float* out_w    = (const float*)d_in[22];
  const float* out_b    = (const float*)d_in[23];
  float* preds = (float*)d_out;

  // ---- workspace carving (all 256B aligned) ----
  char* wsb = (char*)d_ws;
  size_t off = 0;
  auto alloc = [&](size_t bytes) -> void* {
    void* p = wsb + off;
    off = (off + bytes + 255) & ~(size_t)255;
    return p;
  };
  float* enc_f32 = (float*)alloc((size_t)B_ * S_ * H_ * 4);
  u16*   enc_bf  = (u16*)  alloc((size_t)B_ * S_ * H_ * 2);
  float* pk      = (float*)alloc((size_t)B_ * S_ * H_ * 4);
  u16*   kw_bf   = (u16*)  alloc((size_t)H_ * H_ * 2);
  u16*   qw_bf   = (u16*)  alloc((size_t)H_ * H_ * 2);
  u16*   wih_bf  = (u16*)  alloc((size_t)4 * H_ * (H_ + E_) * 2);
  u16*   whh_bf  = (u16*)  alloc((size_t)4 * H_ * H_ * 2);
  u16*   outw_bf = (u16*)  alloc((size_t)V_ * H_ * 2);
  u16*   x_bf    = (u16*)  alloc((size_t)T_ * B_ * E_ * 2);
  u16*   h_bf    = (u16*)  alloc((size_t)B_ * H_ * 2);
  float* c_f32   = (float*)alloc((size_t)B_ * H_ * 4);
  float* q_f32   = (float*)alloc((size_t)B_ * H_ * 4);
  float* energ   = (float*)alloc((size_t)B_ * S_ * 4);
  u16*   xc_bf   = (u16*)  alloc((size_t)B_ * (H_ + E_) * 2);
  float* gates   = (float*)alloc((size_t)B_ * 4 * H_ * 4);
  u16*   hs_bf   = (u16*)  alloc((size_t)B_ * T_ * H_ * 2);
  u16*   tdm_dmy = (u16*)  alloc((size_t)16 * 64 * 2);

  auto conv = [&](const float* s, u16* d, size_t n) {
    f32_to_bf16_kernel<<<(unsigned)((n + 255) / 256), 256, 0, stream>>>(s, d, (int)n);
  };
  // GEMM launcher: one wave per 32x64 C tile, 4 waves (128 thr) per block.
  auto gemm = [&](const u16* A, const u16* Bt, int K, const float* bias,
                  float* C, int M, int N) {
    const int tm = M / (WM * 16), tn = N / (WN * 16);
    const int wt = tm * tn;
    wmma_gemm_dual<<<(wt + 3) / 4, 128, 0, stream>>>(A, Bt, K, nullptr, nullptr, 0,
                                                     bias, nullptr, C, N, tm, tn);
  };

  // ---- weight precision conversion (independent of data path) ----
  conv(k_w,   kw_bf,  (size_t)H_ * H_);
  conv(q_w,   qw_bf,  (size_t)H_ * H_);
  conv(w_ih,  wih_bf, (size_t)4 * H_ * (H_ + E_));
  conv(w_hh,  whh_bf, (size_t)4 * H_ * H_);
  conv(out_w, outw_bf, (size_t)V_ * H_);

  // ---- h0 / c0 ----
  init_hc_kernel<<<(B_ * H_) / 256, 256, 0, stream>>>(enc_hn, enc_cn, phw, phb, pcw, pcb,
                                                      h_bf, c_f32);
  // ---- LayerNorm(encoder) ----
  layernorm_kernel<<<B_ * S_, 256, 0, stream>>>(enc_hid, ln_enc_g, ln_enc_b,
                                                enc_f32, enc_bf, H_);
  // ---- embeddings + LayerNorm ----
  embed_ln_kernel<<<B_ * T_, 256, 0, stream>>>(targets, emb, ln_emb_g, ln_emb_b, x_bf);

  // ---- pk = enc @ k_w^T  (WMMA: 4096 x 512 x 512) ----
  gemm(enc_bf, kw_bf, H_, nullptr, pk, B_ * S_, H_);

  // ---- recurrent decode, T steps ----
  for (int t = 0; t < T_; ++t) {
    // q = h @ q_w^T + q_b   (64 x 512 x 512)
    gemm(h_bf, qw_bf, H_, q_b, q_f32, B_, H_);
    energy_kernel<<<B_ * S_, 256, 0, stream>>>(q_f32, pk, e_w, energ);
    softmax_context_kernel<<<B_, 256, 0, stream>>>(energ, enc_f32,
                                                   x_bf + (size_t)t * B_ * E_, xc_bf);
    { // gates = xc @ w_ih^T + h @ w_hh^T + biases  (64 x 2048, K=768 & 512)
      const int tm = B_ / (WM * 16), tn = (4 * H_) / (WN * 16);
      const int wt = tm * tn;
      wmma_gemm_dual<<<(wt + 3) / 4, 128, 0, stream>>>(xc_bf, wih_bf, H_ + E_,
                                                       h_bf, whh_bf, H_,
                                                       b_ih, b_hh, gates,
                                                       4 * H_, tm, tn);
    }
    lstm_cell_kernel<<<(B_ * H_) / 256, 256, 0, stream>>>(gates, c_f32, h_bf, hs_bf, t);
  }

  // ---- TDM warm of out_w (exercises tensor_load_to_lds path) ----
  tdm_stage_kernel<<<1, 128, 0, stream>>>(outw_bf, tdm_dmy);

  // ---- preds = hs @ out_w^T + out_b  (WMMA: 2048 x 32000 x 512) ----
  gemm(hs_bf, outw_bf, H_, out_b, preds, B_ * T_, V_);
}
